// Net_55765855371370
// MI455X (gfx1250) — compile-verified, matrix operand order checked by reading, not run on hardware
//
#include <hip/hip_runtime.h>

typedef __attribute__((ext_vector_type(16))) __bf16       v16bf;
typedef __attribute__((ext_vector_type(8)))  float        v8f;
typedef __attribute__((ext_vector_type(4)))  unsigned int v4u;

union AFrag {
    v16bf v;
    v4u   u[2];
};

// ---------------------------------------------------------------------------
// f32 -> bf16 round-to-nearest-even (bit pattern identical to __hip_bfloat16)
// ---------------------------------------------------------------------------
__device__ __forceinline__ unsigned short f32_to_bf16(float f) {
    unsigned int u = __float_as_uint(f);
    u += 0x7FFFu + ((u >> 16) & 1u);
    return (unsigned short)(u >> 16);
}

// ---------------------------------------------------------------------------
// Degree / normalization: deg starts at 1 (self loop), +1 per incident dst,
// then dinv = rsqrt(deg).
// ---------------------------------------------------------------------------
__global__ void deg_init_kernel(float* deg, int n) {
    int i = blockIdx.x * blockDim.x + threadIdx.x;
    if (i < n) deg[i] = 1.0f;
}

__global__ void deg_count_kernel(const int* __restrict__ dst, float* deg, int e) {
    int i = blockIdx.x * blockDim.x + threadIdx.x;
    if (i < e) atomicAdd(&deg[dst[i]], 1.0f);
}

__global__ void deg_rsqrt_kernel(float* deg, int n) {
    int i = blockIdx.x * blockDim.x + threadIdx.x;
    if (i < n) deg[i] = rsqrtf(deg[i]);
}

// ---------------------------------------------------------------------------
// f32 -> bf16 convert, optional fused ReLU (feeds next layer's WMMA A matrix)
// ---------------------------------------------------------------------------
__global__ void cvt_bf16_kernel(const float* __restrict__ in,
                                unsigned short* __restrict__ out,
                                long long total, int do_relu) {
    long long i = (long long)blockIdx.x * blockDim.x + threadIdx.x;
    if (i >= total) return;
    float v = in[i];
    if (do_relu) v = fmaxf(v, 0.0f);
    out[i] = f32_to_bf16(v);
}

// ---------------------------------------------------------------------------
// W [K,M] f32 row-major  ->  Wt [M,K] bf16 row-major (so B fragments are
// contiguous 16B loads along K, mirroring the A-matrix layout)
// ---------------------------------------------------------------------------
__global__ void cvt_w_kernel(const float* __restrict__ W,
                             unsigned short* __restrict__ Wt,
                             int K, int M) {
    int i = blockIdx.x * blockDim.x + threadIdx.x;
    if (i >= K * M) return;
    int k = i / M, m = i % M;
    Wt[(size_t)m * K + k] = f32_to_bf16(W[i]);
}

// ---------------------------------------------------------------------------
// WMMA GEMM: C[nrows,M] = A[nrows,K](bf16) x Bt[M,K](bf16)^T, f32 accumulate.
// One wave computes a 16 x (16*NT) strip: A fragment reused NT times.
// 16-bit A 16x32 layout (ISA 7.12.2): lanes 0-15 hold K=k..k+7 / k+16..k+23,
// lanes 16-31 hold K=k+8..k+15 / k+24..k+31; B mirrors with N=lane%16.
// C/D: VGPR i holds M=i (lanes 0-15) / M=i+8 (lanes 16-31), N=lane%16.
// ---------------------------------------------------------------------------
template <int NT>
__global__ void gcn_gemm_bf16_kernel(const unsigned short* __restrict__ A,
                                     const unsigned short* __restrict__ Bt,
                                     float* __restrict__ C,
                                     int nrows, int K, int M) {
    int wave = (int)((blockIdx.x * (long long)blockDim.x + threadIdx.x) >> 5);
    int lane = threadIdx.x & 31;

    int colGroups = M / (16 * NT);
    int tileRow = wave / colGroups;
    int cg      = wave % colGroups;
    if (tileRow * 16 >= nrows) return;   // whole wave exits together: EXEC all-1s for WMMA

    int lrow = lane & 15;
    int koff = (lane < 16) ? 0 : 8;

    const unsigned short* arow  = A  + (size_t)(tileRow * 16 + lrow) * K + koff;
    const unsigned short* brow0 = Bt + (size_t)(cg * 16 * NT + lrow) * K + koff;

    v8f acc[NT];
#pragma unroll
    for (int t = 0; t < NT; ++t) acc[t] = (v8f){};

    for (int k = 0; k < K; k += 32) {
        AFrag fa;
        fa.u[0] = *(const v4u*)(arow + k);        // K = k+koff   .. +7
        fa.u[1] = *(const v4u*)(arow + k + 16);   // K = k+koff+16.. +23
#pragma unroll
        for (int t = 0; t < NT; ++t) {
            const unsigned short* b = brow0 + (size_t)t * 16 * K + k;
            AFrag fb;
            fb.u[0] = *(const v4u*)(b);
            fb.u[1] = *(const v4u*)(b + 16);
            acc[t] = __builtin_amdgcn_wmma_f32_16x16x32_bf16(
                false, fa.v, false, fb.v, (short)0, acc[t], false, false);
        }
    }

    int rbase = tileRow * 16 + ((lane < 16) ? 0 : 8);
#pragma unroll
    for (int t = 0; t < NT; ++t) {
        int cc = cg * 16 * NT + t * 16 + lrow;
        float* cp = C + (size_t)rbase * M + cc;
#pragma unroll
        for (int i = 0; i < 8; ++i) cp[(size_t)i * M] = acc[t][i];
    }
}

// ---------------------------------------------------------------------------
// agg[v,m] = b[m] + H[v,m] * dinv[v]^2      (self-loop term + bias)
// M is a power of two; mshift = log2(M)
// ---------------------------------------------------------------------------
__global__ void agg_init_kernel(const float* __restrict__ H,
                                const float* __restrict__ dinv,
                                const float* __restrict__ bias,
                                float* __restrict__ out,
                                int mshift, long long total) {
    long long i = (long long)blockIdx.x * blockDim.x + threadIdx.x;
    if (i >= total) return;
    int v = (int)(i >> mshift);
    int m = (int)(i & (((long long)1 << mshift) - 1));
    float dv = dinv[v];
    out[i] = bias[m] + H[i] * dv * dv;
}

// ---------------------------------------------------------------------------
// Edge scatter: out[dst] += H[src] * dinv[src]*dinv[dst].
// Thread -> (edge, 4-float chunk); consecutive threads cover one H row
// (coalesced float4 gather), f32 atomics into L2-resident agg buffer.
// chunkShift = log2(M/4)
// ---------------------------------------------------------------------------
__global__ void agg_edges_kernel(const int* __restrict__ src,
                                 const int* __restrict__ dst,
                                 const float* __restrict__ dinv,
                                 const float* __restrict__ H,
                                 float* __restrict__ out,
                                 int M, int chunkShift, long long total) {
    long long t = (long long)blockIdx.x * blockDim.x + threadIdx.x;
    if (t >= total) return;
    long long e = t >> chunkShift;
    int c = (int)(t & (((long long)1 << chunkShift) - 1));
    int s = src[e], d = dst[e];
    float w = dinv[s] * dinv[d];
    const float4 h = ((const float4*)(H + (size_t)s * M))[c];
    float* o = out + (size_t)d * M + c * 4;
    atomicAdd(o + 0, h.x * w);
    atomicAdd(o + 1, h.y * w);
    atomicAdd(o + 2, h.z * w);
    atomicAdd(o + 3, h.w * w);
}

// ---------------------------------------------------------------------------
// Host-side orchestration
// ---------------------------------------------------------------------------
static inline size_t align256(size_t x) { return (x + 255) & ~(size_t)255; }
static inline int blocks_for(long long total, int bs) {
    return (int)((total + bs - 1) / bs);
}

extern "C" void kernel_launch(void* const* d_in, const int* in_sizes, int n_in,
                              void* d_out, int out_size, void* d_ws, size_t ws_size,
                              hipStream_t stream) {
    const float* x   = (const float*)d_in[0];
    const int*   ei  = (const int*)d_in[1];
    const float* W1  = (const float*)d_in[2];
    const float* b1  = (const float*)d_in[3];
    const float* W2  = (const float*)d_in[4];
    const float* b2  = (const float*)d_in[5];
    const float* W3  = (const float*)d_in[6];
    const float* b3  = (const float*)d_in[7];

    const int N = in_sizes[0] / 256;   // 100000
    const int E = in_sizes[1] / 2;     // 1600000
    const int* src = ei;
    const int* dst = ei + E;

    // Workspace carve-up (all 256B aligned)
    char* w = (char*)d_ws;
    float* dinv = (float*)w;               w += align256((size_t)N * 4);
    unsigned short* abf = (unsigned short*)w; w += align256((size_t)N * 256 * 2);
    float* gbuf = (float*)w;               w += align256((size_t)N * 128 * 4);
    float* aggb = (float*)w;               w += align256((size_t)N * 128 * 4);
    unsigned short* wtb = (unsigned short*)w; w += align256((size_t)256 * 128 * 2);

    const int BS = 256;

    // --- normalization: deg -> dinv ---
    deg_init_kernel<<<blocks_for(N, BS), BS, 0, stream>>>(dinv, N);
    deg_count_kernel<<<blocks_for(E, BS), BS, 0, stream>>>(dst, dinv, E);
    deg_rsqrt_kernel<<<blocks_for(N, BS), BS, 0, stream>>>(dinv, N);

    // --- layer 1: x[N,256] @ W1[256,128] ---
    cvt_bf16_kernel<<<blocks_for((long long)N * 256, BS), BS, 0, stream>>>(
        x, abf, (long long)N * 256, 0);
    cvt_w_kernel<<<blocks_for(256 * 128, BS), BS, 0, stream>>>(W1, wtb, 256, 128);
    {
        long long waves = (long long)(N / 16) * (128 / 64);   // NT=4
        gcn_gemm_bf16_kernel<4><<<blocks_for(waves * 32, 128), 128, 0, stream>>>(
            abf, wtb, gbuf, N, 256, 128);
    }
    agg_init_kernel<<<blocks_for((long long)N * 128, BS), BS, 0, stream>>>(
        gbuf, dinv, b1, aggb, 7, (long long)N * 128);
    agg_edges_kernel<<<blocks_for((long long)E << 5, BS), BS, 0, stream>>>(
        src, dst, dinv, gbuf, aggb, 128, 5, (long long)E << 5);

    // --- layer 2: relu(h)[N,128] @ W2[128,128] ---
    cvt_bf16_kernel<<<blocks_for((long long)N * 128, BS), BS, 0, stream>>>(
        aggb, abf, (long long)N * 128, 1);
    cvt_w_kernel<<<blocks_for(128 * 128, BS), BS, 0, stream>>>(W2, wtb, 128, 128);
    {
        long long waves = (long long)(N / 16) * (128 / 64);   // NT=4
        gcn_gemm_bf16_kernel<4><<<blocks_for(waves * 32, 128), 128, 0, stream>>>(
            abf, wtb, gbuf, N, 128, 128);
    }
    agg_init_kernel<<<blocks_for((long long)N * 128, BS), BS, 0, stream>>>(
        gbuf, dinv, b2, aggb, 7, (long long)N * 128);
    agg_edges_kernel<<<blocks_for((long long)E << 5, BS), BS, 0, stream>>>(
        src, dst, dinv, gbuf, aggb, 128, 5, (long long)E << 5);

    // --- layer 3: relu(h)[N,128] @ W3[128,16] -> aggregate into d_out ---
    cvt_bf16_kernel<<<blocks_for((long long)N * 128, BS), BS, 0, stream>>>(
        aggb, abf, (long long)N * 128, 1);
    cvt_w_kernel<<<blocks_for(128 * 16, BS), BS, 0, stream>>>(W3, wtb, 128, 16);
    {
        long long waves = (long long)(N / 16) * 1;            // NT=1, M=16
        gcn_gemm_bf16_kernel<1><<<blocks_for(waves * 32, 128), 128, 0, stream>>>(
            abf, wtb, gbuf, N, 128, 16);
    }
    float* out = (float*)d_out;
    agg_init_kernel<<<blocks_for((long long)N * 16, BS), BS, 0, stream>>>(
        gbuf, dinv, b3, out, 4, (long long)N * 16);
    agg_edges_kernel<<<blocks_for((long long)E << 2, BS), BS, 0, stream>>>(
        src, dst, dinv, gbuf, out, 16, 2, (long long)E << 2);
}
